// StackLSTM_65171833750281
// MI455X (gfx1250) — compile-verified, matrix operand order checked by reading, not run on hardware
//
#include <hip/hip_runtime.h>

// ---------------------------------------------------------------------------
// 2-layer LSTM scan for MI455X (gfx1250, wave32, WMMA).
//
// Per timestep, per layer:  gates[B,4H] = x@W_ih^T + h@W_hh^T + b
// B=128, H=2048, D_IN=1024 -> per-step GEMMs [128 x 8192 x (1024|2048)].
// bf16 weights (112 MB) fit in the 192 MB L2 -> weight-stationary-in-L2,
// v_wmma_f32_16x16x32_bf16 with fp32 accumulation.
// ---------------------------------------------------------------------------

typedef __attribute__((ext_vector_type(16))) __bf16 bf16x16;
typedef __attribute__((ext_vector_type(8)))  float  f32x8;

#define T_STEPS 512
#define BATCH   128
#define DIN     1024
#define H       2048
#define NG      (4 * H)   // 8192 gate columns

// ---- fp32 -> bf16 bulk conversion (grid-stride) ---------------------------
__global__ void cvt_f32_to_bf16(const float* __restrict__ src,
                                __bf16* __restrict__ dst, long n) {
  long i = (long)blockIdx.x * blockDim.x + threadIdx.x;
  long stride = (long)gridDim.x * blockDim.x;
  for (; i < n; i += stride) dst[i] = (__bf16)src[i];
}

// ---- initial state: bf16 h mirrors + fp32 c copies for both layers --------
__global__ void init_state(const float* __restrict__ h0, const float* __restrict__ c0,
                           __bf16* __restrict__ h0b, __bf16* __restrict__ h1b,
                           float* __restrict__ c0f, float* __restrict__ c1f) {
  int i = blockIdx.x * blockDim.x + threadIdx.x;
  if (i < BATCH * H) {
    h0b[i] = (__bf16)h0[i];
    h1b[i] = (__bf16)h0[BATCH * H + i];
    c0f[i] = c0[i];
    c1f[i] = c0[BATCH * H + i];
  }
}

// ---- accumulate A[M,K] x W[N,K]^T into two 16x16 WMMA accumulators --------
// K order chosen here, same for A and B (not the ISA table order): lane holds row (lane&15), 16
// contiguous K values at K-offset (lane>>4)*16 -> plain 32B contiguous loads.
__device__ __forceinline__ void mm_accum(const __bf16* __restrict__ A,
                                         const __bf16* __restrict__ W,
                                         int K, int m0, int n0, int lane,
                                         f32x8* acc) {
  const int r  = lane & 15;
  const int kh = (lane >> 4) * 16;
  const __bf16* ap  = A + (long)(m0 + r) * K + kh;
  const __bf16* wp0 = W + (long)(n0 + r) * K + kh;
  const __bf16* wp1 = W + (long)(n0 + 16 + r) * K + kh;
#pragma unroll 4
  for (int k = 0; k < K; k += 32) {
    // stream-ahead hint for the weight rows (lowers to global_prefetch_b8)
    __builtin_prefetch(wp0 + k + 128, 0, 1);
    __builtin_prefetch(wp1 + k + 128, 0, 1);
    bf16x16 a  = *(const bf16x16*)(ap  + k);
    bf16x16 b0 = *(const bf16x16*)(wp0 + k);
    bf16x16 b1 = *(const bf16x16*)(wp1 + k);
    acc[0] = __builtin_amdgcn_wmma_f32_16x16x32_bf16(
        false, a, false, b0, (short)0, acc[0], false, false);
    acc[1] = __builtin_amdgcn_wmma_f32_16x16x32_bf16(
        false, a, false, b1, (short)0, acc[1], false, false);
  }
}

// ---- fused dual GEMM: gates = A1@W1^T + A2@W2^T + bias --------------------
// grid = (NG/128, BATCH/32), block = 256 (8 waves).
// 8 waves arranged 2(M) x 4(N); each wave owns a 16x32 output tile.
__global__ __launch_bounds__(256)
void gates_gemm(const __bf16* __restrict__ A1, const __bf16* __restrict__ W1, int K1,
                const __bf16* __restrict__ A2, const __bf16* __restrict__ W2, int K2,
                const float* __restrict__ bias, float* __restrict__ gates) {
  const int lane = threadIdx.x & 31;
  const int wave = threadIdx.x >> 5;
  const int wm = wave >> 2;   // 0..1
  const int wn = wave & 3;    // 0..3
  const int m0 = blockIdx.y * 32 + wm * 16;
  const int n0 = blockIdx.x * 128 + wn * 32;

  f32x8 acc[2] = {};
  mm_accum(A1, W1, K1, m0, n0, lane, acc);   // input contribution
  mm_accum(A2, W2, K2, m0, n0, lane, acc);   // recurrent contribution

  // C/D layout: element r of acc -> row m0 + r + (lane>=16 ? 8 : 0),
  //             col n0 + (lane&15)  (acc[1]: +16 cols). Bias is per-column.
  const int col   = n0 + (lane & 15);
  const int rbase = m0 + ((lane >> 4) ? 8 : 0);
  const float bv0 = bias[col];
  const float bv1 = bias[col + 16];
#pragma unroll
  for (int r = 0; r < 8; ++r) {
    gates[(long)(rbase + r) * NG + col]      = acc[0][r] + bv0;
    gates[(long)(rbase + r) * NG + col + 16] = acc[1][r] + bv1;
  }
}

// ---- LSTM pointwise: c' = sig(f)*c + sig(i)*tanh(g); h' = sig(o)*tanh(c') -
__global__ void lstm_pointwise(const float* __restrict__ gates,
                               float* __restrict__ c,
                               __bf16* __restrict__ h_bf,
                               float* __restrict__ h_f32_or_null) {
  int i = blockIdx.x * blockDim.x + threadIdx.x;
  if (i >= BATCH * H) return;
  int b = i / H, n = i - b * H;
  const float* g = gates + (long)b * NG;
  float ig = g[n];
  float fg = g[H + n];
  float gg = g[2 * H + n];
  float og = g[3 * H + n];
  float is = 1.f / (1.f + __expf(-ig));
  float fs = 1.f / (1.f + __expf(-fg));
  float gt = tanhf(gg);
  float os = 1.f / (1.f + __expf(-og));
  float cn = fs * c[i] + is * gt;
  c[i] = cn;
  float hn = os * tanhf(cn);
  h_bf[i] = (__bf16)hn;
  if (h_f32_or_null) h_f32_or_null[i] = hn;
}

// ---------------------------------------------------------------------------
extern "C" void kernel_launch(void* const* d_in, const int* in_sizes, int n_in,
                              void* d_out, int out_size, void* d_ws, size_t ws_size,
                              hipStream_t stream) {
  (void)in_sizes; (void)n_in; (void)out_size; (void)ws_size;
  const float* input_seq = (const float*)d_in[0];   // [T, B, DIN]
  const float* h0        = (const float*)d_in[1];   // [2, B, H]
  const float* c0        = (const float*)d_in[2];   // [2, B, H]
  const float* Wih0      = (const float*)d_in[3];   // [4H, DIN]
  const float* Whh0      = (const float*)d_in[4];   // [4H, H]
  const float* b0        = (const float*)d_in[5];   // [4H]
  const float* Wih1      = (const float*)d_in[6];   // [4H, H]
  const float* Whh1      = (const float*)d_in[7];   // [4H, H]
  const float* b1        = (const float*)d_in[8];   // [4H]

  char* ws = (char*)d_ws;
  size_t off = 0;
  auto take = [&](size_t bytes) -> void* {
    void* p = ws + off;
    off += (bytes + 255) & ~(size_t)255;
    return p;
  };
  __bf16* Wih0b = (__bf16*)take((size_t)NG * DIN * 2);
  __bf16* Whh0b = (__bf16*)take((size_t)NG * H * 2);
  __bf16* Wih1b = (__bf16*)take((size_t)NG * H * 2);
  __bf16* Whh1b = (__bf16*)take((size_t)NG * H * 2);
  __bf16* Xb    = (__bf16*)take((size_t)T_STEPS * BATCH * DIN * 2);
  __bf16* h0b   = (__bf16*)take((size_t)BATCH * H * 2);
  __bf16* h1b   = (__bf16*)take((size_t)BATCH * H * 2);
  float*  c0f   = (float*)take((size_t)BATCH * H * 4);
  float*  c1f   = (float*)take((size_t)BATCH * H * 4);
  float*  gates = (float*)take((size_t)BATCH * NG * 4);

  // One-shot fp32 -> bf16 of weights + the whole input sequence (HBM pass).
  cvt_f32_to_bf16<<<4096, 256, 0, stream>>>(Wih0, Wih0b, (long)NG * DIN);
  cvt_f32_to_bf16<<<4096, 256, 0, stream>>>(Whh0, Whh0b, (long)NG * H);
  cvt_f32_to_bf16<<<4096, 256, 0, stream>>>(Wih1, Wih1b, (long)NG * H);
  cvt_f32_to_bf16<<<4096, 256, 0, stream>>>(Whh1, Whh1b, (long)NG * H);
  cvt_f32_to_bf16<<<8192, 256, 0, stream>>>(input_seq, Xb, (long)T_STEPS * BATCH * DIN);
  init_state<<<(BATCH * H + 255) / 256, 256, 0, stream>>>(h0, c0, h0b, h1b, c0f, c1f);

  const dim3 ggrid(NG / 128, BATCH / 32);   // 64 x 4 = 256 workgroups
  const int pgrid = (BATCH * H + 255) / 256;

  for (int t = 0; t < T_STEPS; ++t) {
    const __bf16* xt = Xb + (size_t)t * BATCH * DIN;
    // Layer 0
    gates_gemm<<<ggrid, 256, 0, stream>>>(xt, Wih0b, DIN, h0b, Whh0b, H, b0, gates);
    lstm_pointwise<<<pgrid, 256, 0, stream>>>(gates, c0f, h0b, nullptr);
    // Layer 1 (input = layer-0 hidden just produced)
    gates_gemm<<<ggrid, 256, 0, stream>>>(h0b, Wih1b, H, h1b, Whh1b, H, b1, gates);
    lstm_pointwise<<<pgrid, 256, 0, stream>>>(
        gates, c1f, h1b, (t == T_STEPS - 1) ? (float*)d_out : nullptr);
  }
}
